// HouseCritic_3401614099017
// MI455X (gfx1250) — matrix operand order repaired and checked by reading, not compiled, not run on hardware
//
#include <hip/hip_runtime.h>
#include <math.h>

typedef __attribute__((ext_vector_type(16))) _Float16 v16h;
typedef __attribute__((ext_vector_type(8)))  float    v8f;

#define NCOLL 200000
#define MNODES 2048
constexpr int NT_COLL = NCOLL / 16;   // 12500 row tiles
constexpr int NT_M    = MNODES / 16;  // 128 row tiles

// ---------------- workspace layout (float offsets) ----------------
constexpr long long O_UP   = 0;                  // 64   u_p
constexpr long long O_US   = O_UP + 64;          // 64   u_s
constexpr long long O_UW   = O_US + 64;          // 32   u_w
constexpr long long O_HP   = O_UW + 32;          // 32   h_p
constexpr long long O_HV   = O_HP + 32;          // 32   h_v
constexpr long long O_HR   = O_HV + 32;          // 32   h_r
constexpr long long O_CL   = O_HR + 32;          // 1    const part of colleague logit
constexpr long long O_GMAX = O_CL + 1;           // 1    global max of logits
constexpr long long O_WSUM = O_GMAX + 1;         // 1    sum of exp-weights (atomic)
constexpr long long O_ACC  = O_WSUM + 1;         // 128  weighted colleague sum (atomic)
constexpr long long O_XU   = O_ACC + 128;        // 128  fused user embedding
constexpr long long O_W1   = O_XU + 128;         // 14336 hyper w1 (224x64)
constexpr long long O_W2   = O_W1 + 14336;       // 2048  hyper w2 (64x32)
constexpr long long O_W3   = O_W2 + 2048;        // 32    hyper w3
constexpr long long O_LOG  = O_W3 + 32;          // 200000 colleague logits
constexpr long long O_PMAX = O_LOG + NCOLL;      // 256  block-partial maxima
constexpr long long O_GX   = O_PMAX + 256;       // 2048*128 gx (row major)
constexpr long long O_HT1  = O_GX + 262144;      // 4*32*2048 GAT1 h, [head][col][row]
constexpr long long O_F1H  = O_HT1 + 262144;     // 4*2048
constexpr long long O_F2H  = O_F1H + 8192;       // 4*2048
constexpr long long O_RMX1 = O_F2H + 8192;       // 4*2048
constexpr long long O_RSM1 = O_RMX1 + 8192;      // 4*2048
constexpr long long O_HCAT = O_RSM1 + 8192;      // 2048*128 (row major)
constexpr long long O_HT2  = O_HCAT + 262144;    // 128*2048 GAT2 h, [col][row]
constexpr long long O_F1B  = O_HT2 + 262144;     // 2048
constexpr long long O_F2B  = O_F1B + 2048;       // 2048
constexpr long long O_RMX2 = O_F2B + 2048;       // 2048
constexpr long long O_RSM2 = O_RMX2 + 2048;      // 2048
constexpr long long O_NEIGH= O_RSM2 + 2048;      // 128 column sums of gat2 out (atomic)

// ---------------- helpers ----------------
__device__ __forceinline__ float celuf(float x) { return x > 0.f ? x : expm1f(x); }
__device__ __forceinline__ float eluf (float x) { return x > 0.f ? x : expm1f(x); }
__device__ __forceinline__ float lrelu(float x) { return x > 0.f ? x : 0.2f * x; }

__device__ __forceinline__ v8f vzero8() {
  v8f z;
#pragma unroll
  for (int i = 0; i < 8; ++i) z[i] = 0.f;
  return z;
}

__device__ __forceinline__ v8f wmma32(v16h a, v16h b, v8f c) {
  // D = A(16x32 f16) * B(32x16 f16) + C(16x16 f32)
  return __builtin_amdgcn_wmma_f32_16x16x32_f16(false, a, false, b, (short)0, c,
                                                false, false);
}

// A fragment: rowk points at X[m][kk]; lane covers K = h8+{0..7} and h8+16..23
__device__ __forceinline__ v16h load_a_row(const float* __restrict__ rowk, int h8) {
  v16h a;
#pragma unroll
  for (int e = 0; e < 8; ++e) {
    a[e]     = (_Float16)rowk[h8 + e];
    a[e + 8] = (_Float16)rowk[h8 + 16 + e];
  }
  return a;
}

// B fragment from row-major K x N weight matrix
__device__ __forceinline__ v16h load_b_g(const float* __restrict__ W, int ldn,
                                         int kk, int ncol, int lane) {
  int n  = ncol + (lane & 15);
  int k0 = kk + ((lane >> 4) << 3);
  const float* p = W + (long long)k0 * ldn + n;
  v16h b;
#pragma unroll
  for (int e = 0; e < 16; ++e) b[e] = (_Float16)p[(e < 8 ? e : e + 8) * ldn];
  return b;
}

// B fragment from transposed storage hT[col][row] (contiguous in K)
__device__ __forceinline__ v16h load_bT(const float* __restrict__ hT, int kk,
                                        int ncol, int lane) {
  const float* p = hT + (long long)(ncol + (lane & 15)) * MNODES + kk + ((lane >> 4) << 3);
  v16h b;
#pragma unroll
  for (int e = 0; e < 16; ++e) b[e] = (_Float16)p[e < 8 ? e : e + 8];
  return b;
}

// celu(X[16x64] @ W[64x64] + b) for one 16-row tile -> acc[4] in C/D layout
__device__ __forceinline__ void embed_tile64(const float* __restrict__ Xbase,
                                             const float* __restrict__ W,
                                             const float* __restrict__ bias,
                                             v8f acc[4], int lane) {
  int m = lane & 15, h8 = (lane >> 4) * 8;
#pragma unroll
  for (int nt = 0; nt < 4; ++nt) acc[nt] = vzero8();
#pragma unroll
  for (int kk = 0; kk < 64; kk += 32) {
    v16h a = load_a_row(Xbase + m * 64 + kk, h8);
#pragma unroll
    for (int nt = 0; nt < 4; ++nt) {
      v16h b = load_b_g(W, 64, kk, nt * 16, lane);
      acc[nt] = wmma32(a, b, acc[nt]);
    }
  }
  int n15 = lane & 15;
#pragma unroll
  for (int nt = 0; nt < 4; ++nt) {
    float bb = bias[nt * 16 + n15];
#pragma unroll
    for (int r = 0; r < 8; ++r) {
      float v = acc[nt][r] + bb;
      acc[nt][r] = v > 0.f ? v : expm1f(v);
    }
  }
}

// ---------------- kernels ----------------
__global__ void kinit(float* ws) {
  int t = threadIdx.x;
  if (t < 128) { ws[O_ACC + t] = 0.f; ws[O_NEIGH + t] = 0.f; }
  if (t == 0)  ws[O_WSUM] = 0.f;
}

__global__ void k_user_house(const float* xup, const float* xus, const float* xuw,
                             const float* xhp, const float* xhv, const float* xhr,
                             const float* Wup, const float* bup,
                             const float* Wus, const float* bus,
                             const float* Wuw, const float* buw,
                             const float* Whp, const float* bhp,
                             const float* Whv, const float* bhv,
                             const float* Whr, const float* bhr,
                             const float* vw, float* ws) {
  int t = threadIdx.x;
  if (t < 64)       { float s = bup[t];      for (int k = 0; k < 64; ++k) s += xup[k] * Wup[k * 64 + t];        ws[O_UP + t]       = celuf(s); }
  else if (t < 128) { int e = t - 64;  float s = bus[e]; for (int k = 0; k < 64; ++k) s += xus[k] * Wus[k * 64 + e]; ws[O_US + e] = celuf(s); }
  else if (t < 160) { int e = t - 128; float s = buw[e]; for (int k = 0; k < 32; ++k) s += xuw[k] * Wuw[k * 32 + e]; ws[O_UW + e] = celuf(s); }
  else if (t < 192) { int e = t - 160; float s = bhp[e]; for (int k = 0; k < 32; ++k) s += xhp[k] * Whp[k * 32 + e]; ws[O_HP + e] = celuf(s); }
  else if (t < 224) { int e = t - 192; float s = bhv[e]; for (int k = 0; k < 32; ++k) s += xhv[k] * Whv[k * 32 + e]; ws[O_HV + e] = celuf(s); }
  else              { int e = t - 224; float s = bhr[e]; for (int k = 0; k < 32; ++k) s += xhr[k] * Whr[k * 32 + e]; ws[O_HR + e] = celuf(s); }
  __syncthreads();
  if (t == 0) {
    float c = 0.f;
    for (int j = 0; j < 64; ++j) c += ws[O_US + j] * vw[j] + ws[O_UP + j] * vw[64 + j];
    ws[O_CL] = c;
  }
}

// pass 1: colleague embeddings (WMMA) -> per-row attention logits
__global__ void k_coll_logits(const float* __restrict__ xp, const float* __restrict__ xs,
                              const float* __restrict__ Wup, const float* __restrict__ bup,
                              const float* __restrict__ Wus, const float* __restrict__ bus,
                              const float* __restrict__ vw, float* __restrict__ ws) {
  int lane = threadIdx.x & 31, wave = threadIdx.x >> 5;
  int tile = blockIdx.x * 8 + wave;
  if (tile >= NT_COLL) return;
  long long base = (long long)tile * 16;
  v8f accP[4], accS[4];
  embed_tile64(xp + base * 64, Wup, bup, accP, lane);
  embed_tile64(xs + base * 64, Wus, bus, accS, lane);
  int n15 = lane & 15, h = lane >> 4;
  float r8[8];
#pragma unroll
  for (int r = 0; r < 8; ++r) r8[r] = 0.f;
#pragma unroll
  for (int nt = 0; nt < 4; ++nt) {
    float vwp = vw[128 + nt * 16 + n15];
    float vws = vw[192 + nt * 16 + n15];
#pragma unroll
    for (int r = 0; r < 8; ++r) r8[r] += accP[nt][r] * vwp + accS[nt][r] * vws;
  }
#pragma unroll
  for (int r = 0; r < 8; ++r) {
    float v = r8[r];
    for (int mm = 1; mm < 16; mm <<= 1) v += __shfl_xor(v, mm, 32);
    r8[r] = v;
  }
  if (n15 == 0) {
    float cl = ws[O_CL];
#pragma unroll
    for (int r = 0; r < 8; ++r) ws[O_LOG + base + h * 8 + r] = cl + r8[r];
  }
}

__global__ void k_max1(float* ws) {
  __shared__ float red[256];
  const float* lg = ws + O_LOG;
  float m = -3.4e38f;
  for (int i = blockIdx.x * 256 + threadIdx.x; i < NCOLL; i += 256 * 256)
    m = fmaxf(m, lg[i]);
  red[threadIdx.x] = m; __syncthreads();
  for (int s = 128; s > 0; s >>= 1) {
    if (threadIdx.x < s) red[threadIdx.x] = fmaxf(red[threadIdx.x], red[threadIdx.x + s]);
    __syncthreads();
  }
  if (threadIdx.x == 0) ws[O_PMAX + blockIdx.x] = red[0];
}

__global__ void k_max2(float* ws) {
  __shared__ float red[256];
  red[threadIdx.x] = ws[O_PMAX + threadIdx.x]; __syncthreads();
  for (int s = 128; s > 0; s >>= 1) {
    if (threadIdx.x < s) red[threadIdx.x] = fmaxf(red[threadIdx.x], red[threadIdx.x + s]);
    __syncthreads();
  }
  if (threadIdx.x == 0) ws[O_GMAX] = red[0];
}

// pass 2: recompute embeddings, accumulate softmax-weighted colleague vector
__global__ void k_accum(const float* __restrict__ xp, const float* __restrict__ xs,
                        const float* __restrict__ Wup, const float* __restrict__ bup,
                        const float* __restrict__ Wus, const float* __restrict__ bus,
                        float* __restrict__ ws) {
  __shared__ float blkacc[128];
  __shared__ float blkw;
  if (threadIdx.x < 128) blkacc[threadIdx.x] = 0.f;
  if (threadIdx.x == 128) blkw = 0.f;
  __syncthreads();
  int lane = threadIdx.x & 31, wave = threadIdx.x >> 5;
  int tile = blockIdx.x * 8 + wave;
  if (tile < NT_COLL) {
    long long base = (long long)tile * 16;
    v8f accP[4], accS[4];
    embed_tile64(xp + base * 64, Wup, bup, accP, lane);
    embed_tile64(xs + base * 64, Wus, bus, accS, lane);
    int h = lane >> 4;
    float gm = ws[O_GMAX];
    float w8[8];
#pragma unroll
    for (int r = 0; r < 8; ++r) w8[r] = expf(ws[O_LOG + base + h * 8 + r] - gm);
    float pc[8];
#pragma unroll
    for (int i = 0; i < 8; ++i) pc[i] = 0.f;
#pragma unroll
    for (int nt = 0; nt < 4; ++nt)
#pragma unroll
      for (int r = 0; r < 8; ++r) {
        pc[nt]     += w8[r] * accP[nt][r];
        pc[4 + nt] += w8[r] * accS[nt][r];
      }
#pragma unroll
    for (int i = 0; i < 8; ++i) pc[i] += __shfl_xor(pc[i], 16, 32);
    float wsl = 0.f;
#pragma unroll
    for (int r = 0; r < 8; ++r) wsl += w8[r];
    wsl += __shfl_xor(wsl, 16, 32);
    if (lane < 16) {
#pragma unroll
      for (int nt = 0; nt < 4; ++nt) {
        atomicAdd(&blkacc[nt * 16 + lane], pc[nt]);
        atomicAdd(&blkacc[64 + nt * 16 + lane], pc[4 + nt]);
      }
    }
    if (lane == 0) atomicAdd(&blkw, wsl);
  }
  __syncthreads();
  if (threadIdx.x < 128) atomicAdd(&ws[O_ACC + threadIdx.x], blkacc[threadIdx.x]);
  if (threadIdx.x == 128) atomicAdd(&ws[O_WSUM], blkw);
}

// user fusion MLP: fu(288) -> 256 -> 128
__global__ void k_fuse(const float* Wf1, const float* bf1,
                       const float* Wf2, const float* bf2, float* ws) {
  __shared__ float fu[288];
  __shared__ float x1[256];
  int t = threadIdx.x;
  float inv = 1.0f / (ws[O_WSUM] * (float)NCOLL);
  if (t < 64)       fu[t] = ws[O_UP + t];
  else if (t < 128) fu[t] = ws[O_US + t - 64];
  else if (t < 160) fu[t] = ws[O_UW + t - 128];
  else              fu[t] = ws[O_ACC + (t - 160)] * inv;
  if (t < 32)       fu[256 + t] = ws[O_ACC + 96 + t] * inv;
  __syncthreads();
  { float s = bf1[t]; for (int k = 0; k < 288; ++k) s += fu[k] * Wf1[k * 256 + t]; x1[t] = celuf(s); }
  __syncthreads();
  if (t < 128) { float s = bf2[t]; for (int k = 0; k < 256; ++k) s += x1[k] * Wf2[k * 128 + t]; ws[O_XU + t] = celuf(s); }
}

// hypernetwork generated weights
__global__ void k_hyper(const float* Wm1, const float* bm1,
                        const float* Wm2, const float* bm2,
                        const float* Wm3, const float* bm3, float* ws) {
  int i = blockIdx.x * 256 + threadIdx.x;
  const float* xu = ws + O_XU;
  if (i < 14336) {
    float s = bm1[i]; for (int k = 0; k < 128; ++k) s += xu[k] * Wm1[k * 14336 + i];
    ws[O_W1 + i] = celuf(s);
  } else if (i < 16384) {
    int j = i - 14336;
    float s = bm2[j]; for (int k = 0; k < 128; ++k) s += xu[k] * Wm2[k * 2048 + j];
    ws[O_W2 + j] = celuf(s);
  } else if (i < 16416) {
    int j = i - 16384;
    float s = bm3[j]; for (int k = 0; k < 128; ++k) s += xu[k] * Wm3[k * 32 + j];
    ws[O_W3 + j] = celuf(s);
  }
}

// neighborhood node embeddings gx = [celu(np@Wup+b), celu(ns@Wus+b)]
__global__ void k_gx(const float* __restrict__ xnp, const float* __restrict__ xns,
                     const float* __restrict__ Wup, const float* __restrict__ bup,
                     const float* __restrict__ Wus, const float* __restrict__ bus,
                     float* __restrict__ ws) {
  int lane = threadIdx.x & 31, wave = threadIdx.x >> 5;
  int tile = blockIdx.x * 8 + wave;     // exactly 128 tiles
  long long base = (long long)tile * 16;
  v8f accP[4], accS[4];
  embed_tile64(xnp + base * 64, Wup, bup, accP, lane);
  embed_tile64(xns + base * 64, Wus, bus, accS, lane);
  int n15 = lane & 15, h = lane >> 4;
  float* gx = ws + O_GX;
#pragma unroll
  for (int nt = 0; nt < 4; ++nt)
#pragma unroll
    for (int r = 0; r < 8; ++r) {
      long long row = base + r + 8 * h;
      int col = nt * 16 + n15;
      gx[row * 128 + col]      = accP[nt][r];
      gx[row * 128 + 64 + col] = accS[nt][r];
    }
}

// GAT layer 1: h = gx @ Wg[head]; f1 = h@ag[:32]; f2 = h@ag[32:]
__global__ void k_gat1h(const float* __restrict__ Wg, const float* __restrict__ ag,
                        float* __restrict__ ws) {
  int lane = threadIdx.x & 31, wave = threadIdx.x >> 5;
  int g = blockIdx.x * 8 + wave;        // 512 waves
  int head = g & 3, tile = g >> 2;
  long long base = (long long)tile * 16;
  const float* gx = ws + O_GX;
  const float* W = Wg + head * 4096;    // [128][32]
  int m = lane & 15, h8 = (lane >> 4) * 8;
  v8f acc[2] = {vzero8(), vzero8()};
#pragma unroll
  for (int kk = 0; kk < 128; kk += 32) {
    v16h a = load_a_row(gx + (base + m) * 128 + kk, h8);
#pragma unroll
    for (int nt = 0; nt < 2; ++nt) {
      v16h b = load_b_g(W, 32, kk, nt * 16, lane);
      acc[nt] = wmma32(a, b, acc[nt]);
    }
  }
  int n15 = lane & 15, h = lane >> 4;
  float* hT = ws + O_HT1 + (long long)head * 32 * MNODES;
  float p1[8], p2[8];
#pragma unroll
  for (int r = 0; r < 8; ++r) { p1[r] = 0.f; p2[r] = 0.f; }
#pragma unroll
  for (int nt = 0; nt < 2; ++nt) {
    int col = nt * 16 + n15;
    float a1 = ag[head * 64 + col];
    float a2 = ag[head * 64 + 32 + col];
#pragma unroll
    for (int r = 0; r < 8; ++r) {
      long long row = base + r + 8 * h;
      float v = acc[nt][r];
      hT[(long long)col * MNODES + row] = v;
      p1[r] += v * a1;
      p2[r] += v * a2;
    }
  }
#pragma unroll
  for (int r = 0; r < 8; ++r) {
    float v1 = p1[r], v2 = p2[r];
    for (int mm = 1; mm < 16; mm <<= 1) { v1 += __shfl_xor(v1, mm, 32); v2 += __shfl_xor(v2, mm, 32); }
    p1[r] = v1; p2[r] = v2;
  }
  if (n15 == 0) {
#pragma unroll
    for (int r = 0; r < 8; ++r) {
      long long row = base + 8 * h + r;
      ws[O_F1H + head * MNODES + row] = p1[r];
      ws[O_F2H + head * MNODES + row] = p2[r];
    }
  }
}

// per-row softmax stats (max & sum) over masked leaky(f1_i + f2_j); H heads
__global__ void k_rowstats(const int* __restrict__ adj,
                           const float* __restrict__ f1, const float* __restrict__ f2,
                           float* __restrict__ rmax, float* __restrict__ rsum, int H) {
  __shared__ int adjr[MNODES];
  __shared__ float red[256];
  int row = blockIdx.x, t = threadIdx.x;
  for (int j = t; j < MNODES; j += 256) adjr[j] = adj[(long long)row * MNODES + j];
  __syncthreads();
  for (int head = 0; head < H; ++head) {
    float f1v = f1[head * MNODES + row];
    const float* f2p = f2 + head * MNODES;
    float mloc = -3.4e38f;
    for (int j = t; j < MNODES; j += 256) {
      float e = adjr[j] > 0 ? lrelu(f1v + f2p[j]) : -1e9f;
      mloc = fmaxf(mloc, e);
    }
    red[t] = mloc; __syncthreads();
    for (int s = 128; s > 0; s >>= 1) {
      if (t < s) red[t] = fmaxf(red[t], red[t + s]);
      __syncthreads();
    }
    float mx = red[0]; __syncthreads();
    float sloc = 0.f;
    for (int j = t; j < MNODES; j += 256) {
      float e = adjr[j] > 0 ? lrelu(f1v + f2p[j]) : -1e9f;
      sloc += expf(e - mx);
    }
    red[t] = sloc; __syncthreads();
    for (int s = 128; s > 0; s >>= 1) {
      if (t < s) red[t] += red[t + s];
      __syncthreads();
    }
    if (t == 0) { rmax[head * MNODES + row] = mx; rsum[head * MNODES + row] = red[0]; }
    __syncthreads();
  }
}

// GAT layer 1 apply: hcat[:, head*32 + c] = elu(att @ h) ; att built on the fly
__global__ void k_gat1apply(const int* __restrict__ adj, float* __restrict__ ws) {
  int lane = threadIdx.x & 31, wave = threadIdx.x >> 5;
  int g = blockIdx.x * 8 + wave;        // 512 waves
  int head = g & 3, tile = g >> 2;
  long long base = (long long)tile * 16;
  int m = lane & 15, h8 = (lane >> 4) * 8;
  long long row = base + m;
  float f1v = ws[O_F1H + head * MNODES + row];
  float mx  = ws[O_RMX1 + head * MNODES + row];
  float inv = 1.0f / ws[O_RSM1 + head * MNODES + row];
  const int* adjrow = adj + row * MNODES;
  const float* f2p = ws + O_F2H + head * MNODES;
  const float* hT  = ws + O_HT1 + (long long)head * 32 * MNODES;
  v8f acc[2] = {vzero8(), vzero8()};
  for (int kk = 0; kk < MNODES; kk += 32) {
    v16h a;
#pragma unroll
    for (int e = 0; e < 16; ++e) {
      int j = kk + h8 + (e < 8 ? e : e + 8);
      float ev = adjrow[j] > 0 ? lrelu(f1v + f2p[j]) : -1e9f;
      a[e] = (_Float16)(expf(ev - mx) * inv);
    }
#pragma unroll
    for (int nt = 0; nt < 2; ++nt) {
      v16h b = load_bT(hT, kk, nt * 16, lane);
      acc[nt] = wmma32(a, b, acc[nt]);
    }
  }
  int n15 = lane & 15, h = lane >> 4;
  float* hcat = ws + O_HCAT;
#pragma unroll
  for (int nt = 0; nt < 2; ++nt)
#pragma unroll
    for (int r = 0; r < 8; ++r) {
      long long wr = base + r + 8 * h;
      int wc = head * 32 + nt * 16 + n15;
      hcat[wr * 128 + wc] = eluf(acc[nt][r]);
    }
}

// GAT layer 2: h2 = hcat @ Wo ; f1b/f2b from ao
__global__ void k_gat2h(const float* __restrict__ Wo, const float* __restrict__ ao,
                        float* __restrict__ ws) {
  int lane = threadIdx.x & 31, wave = threadIdx.x >> 5;
  int tile = blockIdx.x * 8 + wave;     // 128 tiles
  long long base = (long long)tile * 16;
  const float* hcat = ws + O_HCAT;
  int m = lane & 15, h8 = (lane >> 4) * 8;
  v8f acc[8];
#pragma unroll
  for (int nt = 0; nt < 8; ++nt) acc[nt] = vzero8();
#pragma unroll
  for (int kk = 0; kk < 128; kk += 32) {
    v16h a = load_a_row(hcat + (base + m) * 128 + kk, h8);
#pragma unroll
    for (int nt = 0; nt < 8; ++nt) {
      v16h b = load_b_g(Wo, 128, kk, nt * 16, lane);
      acc[nt] = wmma32(a, b, acc[nt]);
    }
  }
  int n15 = lane & 15, h = lane >> 4;
  float* hT2 = ws + O_HT2;
  float p1[8], p2[8];
#pragma unroll
  for (int r = 0; r < 8; ++r) { p1[r] = 0.f; p2[r] = 0.f; }
#pragma unroll
  for (int nt = 0; nt < 8; ++nt) {
    int col = nt * 16 + n15;
    float a1 = ao[col], a2 = ao[128 + col];
#pragma unroll
    for (int r = 0; r < 8; ++r) {
      long long row = base + r + 8 * h;
      float v = acc[nt][r];
      hT2[(long long)col * MNODES + row] = v;
      p1[r] += v * a1;
      p2[r] += v * a2;
    }
  }
#pragma unroll
  for (int r = 0; r < 8; ++r) {
    float v1 = p1[r], v2 = p2[r];
    for (int mm = 1; mm < 16; mm <<= 1) { v1 += __shfl_xor(v1, mm, 32); v2 += __shfl_xor(v2, mm, 32); }
    p1[r] = v1; p2[r] = v2;
  }
  if (n15 == 0) {
#pragma unroll
    for (int r = 0; r < 8; ++r) {
      long long row = base + 8 * h + r;
      ws[O_F1B + row] = p1[r];
      ws[O_F2B + row] = p2[r];
    }
  }
}

// GAT layer 2 apply + column-sum accumulate (for mean over nodes)
__global__ void k_gat2apply(const int* __restrict__ adj, float* __restrict__ ws) {
  __shared__ float blk[128];
  if (threadIdx.x < 128) blk[threadIdx.x] = 0.f;
  __syncthreads();
  int lane = threadIdx.x & 31, wave = threadIdx.x >> 5;
  int tile = blockIdx.x * 8 + wave;     // 128 tiles exactly
  long long base = (long long)tile * 16;
  int m = lane & 15, h8 = (lane >> 4) * 8;
  long long row = base + m;
  float f1v = ws[O_F1B + row];
  float mx  = ws[O_RMX2 + row];
  float inv = 1.0f / ws[O_RSM2 + row];
  const int* adjrow = adj + row * MNODES;
  const float* f2p = ws + O_F2B;
  const float* hT2 = ws + O_HT2;
  v8f acc[8];
#pragma unroll
  for (int nt = 0; nt < 8; ++nt) acc[nt] = vzero8();
  for (int kk = 0; kk < MNODES; kk += 32) {
    v16h a;
#pragma unroll
    for (int e = 0; e < 16; ++e) {
      int j = kk + h8 + (e < 8 ? e : e + 8);
      float ev = adjrow[j] > 0 ? lrelu(f1v + f2p[j]) : -1e9f;
      a[e] = (_Float16)(expf(ev - mx) * inv);
    }
#pragma unroll
    for (int nt = 0; nt < 8; ++nt) {
      v16h b = load_bT(hT2, kk, nt * 16, lane);
      acc[nt] = wmma32(a, b, acc[nt]);
    }
  }
  float pc[8];
#pragma unroll
  for (int nt = 0; nt < 8; ++nt) {
    float s = 0.f;
#pragma unroll
    for (int r = 0; r < 8; ++r) s += eluf(acc[nt][r]);
    pc[nt] = s;
  }
#pragma unroll
  for (int nt = 0; nt < 8; ++nt) pc[nt] += __shfl_xor(pc[nt], 16, 32);
  if (lane < 16) {
#pragma unroll
    for (int nt = 0; nt < 8; ++nt) atomicAdd(&blk[nt * 16 + lane], pc[nt]);
  }
  __syncthreads();
  if (threadIdx.x < 128) atomicAdd(&ws[O_NEIGH + threadIdx.x], blk[threadIdx.x]);
}

// meta MLP with hyper-generated weights -> sigmoid scalar
__global__ void k_final(float* ws, float* out) {
  __shared__ float xh[224];
  __shared__ float z1[64];
  __shared__ float z2[32];
  int t = threadIdx.x;
  if (t < 32) { xh[t] = ws[O_HP + t]; xh[32 + t] = ws[O_HV + t]; xh[64 + t] = ws[O_HR + t]; }
  if (t < 128) xh[96 + t] = ws[O_NEIGH + t] / (float)MNODES;
  __syncthreads();
  if (t < 64) {
    float s = 0.f;
    for (int k = 0; k < 224; ++k) s += xh[k] * ws[O_W1 + k * 64 + t];
    z1[t] = fmaxf(s, 0.f);
  }
  __syncthreads();
  if (t < 32) {
    float s = 0.f;
    for (int k = 0; k < 64; ++k) s += z1[k] * ws[O_W2 + k * 32 + t];
    z2[t] = fmaxf(s, 0.f);
  }
  __syncthreads();
  if (t == 0) {
    float s = 0.f;
    for (int k = 0; k < 32; ++k) s += z2[k] * ws[O_W3 + k];
    out[0] = 1.0f / (1.0f + expf(-s));
  }
}

extern "C" void kernel_launch(void* const* d_in, const int* in_sizes, int n_in,
                              void* d_out, int out_size, void* d_ws, size_t ws_size,
                              hipStream_t stream) {
  (void)in_sizes; (void)n_in; (void)out_size; (void)ws_size;
  const float* xup = (const float*)d_in[0];
  const float* xus = (const float*)d_in[1];
  const float* xuw = (const float*)d_in[2];
  const float* xcp = (const float*)d_in[3];
  const float* xcs = (const float*)d_in[4];
  const float* xhp = (const float*)d_in[5];
  const float* xhv = (const float*)d_in[6];
  const float* xhr = (const float*)d_in[7];
  const float* xnp = (const float*)d_in[8];
  const float* xns = (const float*)d_in[9];
  const int*   adj = (const int*)d_in[10];
  const float* Wup = (const float*)d_in[11]; const float* bup = (const float*)d_in[12];
  const float* Wus = (const float*)d_in[13]; const float* bus = (const float*)d_in[14];
  const float* Wuw = (const float*)d_in[15]; const float* buw = (const float*)d_in[16];
  const float* vw  = (const float*)d_in[17];
  const float* Wf1 = (const float*)d_in[18]; const float* bf1 = (const float*)d_in[19];
  const float* Wf2 = (const float*)d_in[20]; const float* bf2 = (const float*)d_in[21];
  const float* Wm1 = (const float*)d_in[22]; const float* bm1 = (const float*)d_in[23];
  const float* Wm2 = (const float*)d_in[24]; const float* bm2 = (const float*)d_in[25];
  const float* Wm3 = (const float*)d_in[26]; const float* bm3 = (const float*)d_in[27];
  const float* Whp = (const float*)d_in[28]; const float* bhp = (const float*)d_in[29];
  const float* Whv = (const float*)d_in[30]; const float* bhv = (const float*)d_in[31];
  const float* Whr = (const float*)d_in[32]; const float* bhr = (const float*)d_in[33];
  const float* Wg  = (const float*)d_in[34]; const float* ag  = (const float*)d_in[35];
  const float* Wo  = (const float*)d_in[36]; const float* ao  = (const float*)d_in[37];
  float* ws = (float*)d_ws;
  float* out = (float*)d_out;

  kinit<<<1, 256, 0, stream>>>(ws);
  k_user_house<<<1, 256, 0, stream>>>(xup, xus, xuw, xhp, xhv, xhr,
                                      Wup, bup, Wus, bus, Wuw, buw,
                                      Whp, bhp, Whv, bhv, Whr, bhr, vw, ws);
  k_coll_logits<<<(NT_COLL + 7) / 8, 256, 0, stream>>>(xcp, xcs, Wup, bup, Wus, bus, vw, ws);
  k_max1<<<256, 256, 0, stream>>>(ws);
  k_max2<<<1, 256, 0, stream>>>(ws);
  k_accum<<<(NT_COLL + 7) / 8, 256, 0, stream>>>(xcp, xcs, Wup, bup, Wus, bus, ws);
  k_fuse<<<1, 256, 0, stream>>>(Wf1, bf1, Wf2, bf2, ws);
  k_hyper<<<65, 256, 0, stream>>>(Wm1, bm1, Wm2, bm2, Wm3, bm3, ws);
  k_gx<<<NT_M / 8, 256, 0, stream>>>(xnp, xns, Wup, bup, Wus, bus, ws);
  k_gat1h<<<NT_M * 4 / 8, 256, 0, stream>>>(Wg, ag, ws);
  k_rowstats<<<MNODES, 256, 0, stream>>>(adj, ws + O_F1H, ws + O_F2H,
                                         ws + O_RMX1, ws + O_RSM1, 4);
  k_gat1apply<<<NT_M * 4 / 8, 256, 0, stream>>>(adj, ws);
  k_gat2h<<<NT_M / 8, 256, 0, stream>>>(Wo, ao, ws);
  k_rowstats<<<MNODES, 256, 0, stream>>>(adj, ws + O_F1B, ws + O_F2B,
                                         ws + O_RMX2, ws + O_RSM2, 1);
  k_gat2apply<<<NT_M / 8, 256, 0, stream>>>(adj, ws);
  k_final<<<1, 256, 0, stream>>>(ws, out);
}
